// LinearAttention_4853313044970
// MI455X (gfx1250) — matrix operand order repaired and checked by reading, not compiled, not run on hardware
//
#include <hip/hip_runtime.h>

typedef __attribute__((ext_vector_type(8)))  __bf16 v8bf;
typedef __attribute__((ext_vector_type(16))) __bf16 v16bf;
typedef __attribute__((ext_vector_type(8)))  float  v8f;

constexpr int D    = 64;         // head dim
constexpr int CH   = 128;        // chunk length
constexpr int SSEQ = 8192;       // sequence length
constexpr int NCH  = SSEQ / CH;  // 64 chunks per (b,h)
constexpr int BH   = 64;         // B*H
constexpr int SEGG = 16;         // segments per (b,h)
constexpr int SEGC = NCH / SEGG; // chunks per segment (4)

// ln(10000)/32
#define ROPE_LC 0.2878231366242557f

__device__ inline __bf16 tobf(float f) {
  unsigned u = __builtin_bit_cast(unsigned, f);
  u += 0x7FFFu + ((u >> 16) & 1u);          // round-to-nearest-even
  unsigned short h = (unsigned short)(u >> 16);
  return __builtin_bit_cast(__bf16, h);
}

// A fragment: 16x32 bf16 tile, row-major source (rows m0.., cols k0..k0+31), stride ld.
// Per ISA 7.12.2: lanes 0-15 -> M=0..15, K groups {hi*8+0..7, 16+hi*8+0..7}.
__device__ inline v16bf frag_a(const __bf16* base, int ld, int m0, int k0) {
  int lane = threadIdx.x & 31;
  const __bf16* p = base + (m0 + (lane & 15)) * ld + k0 + ((lane >> 4) << 3);
  v8bf lo = *(const v8bf*)p;
  v8bf hi = *(const v8bf*)(p + 16);
  return __builtin_shufflevector(lo, hi, 0,1,2,3,4,5,6,7,8,9,10,11,12,13,14,15);
}

// B fragment: 32x16 bf16 tile B[k][n], supplied via its transpose Bt[n][k]
// (row-major, stride ld). Lane: n = lane&15, K groups {hi*8+0..7, 16+hi*8+0..7}.
__device__ inline v16bf frag_bt(const __bf16* bt, int ld, int n0, int k0) {
  int lane = threadIdx.x & 31;
  const __bf16* p = bt + (n0 + (lane & 15)) * ld + k0 + ((lane >> 4) << 3);
  v8bf lo = *(const v8bf*)p;
  v8bf hi = *(const v8bf*)(p + 16);
  return __builtin_shufflevector(lo, hi, 0,1,2,3,4,5,6,7,8,9,10,11,12,13,14,15);
}

__device__ inline v8f wmma_bf16(v16bf a, v16bf b, v8f c) {
  return __builtin_amdgcn_wmma_f32_16x16x32_bf16(false, a, false, b, (short)0, c,
                                                 false, false);
}

// ---------------------------------------------------------------------------
// RoPE via rotation composition: ang(s,d) = s0*inv[d] + r*inv[d], s0 = c*128.
// Per-thread row rotations (chunk-invariant) kept in registers; per-chunk base
// rotation (32 entries) kept in LDS and advanced by complex multiply with
// rot128[d] = e^{i*128*inv[d]}.  No transcendentals in the chunk loop.
// ---------------------------------------------------------------------------
__device__ inline void rope_init(int c0, float* cr, float* sr,
                                 float2* sRotBase, float2* sRot128) {
  int t = threadIdx.x;
  if (t < 32) {
    float inv = expf(-(float)t * ROPE_LC);
    float sn, cs;
    sincosf((float)(c0 * CH) * inv, &sn, &cs);   // precise: arg up to ~8e3
    sRotBase[t] = make_float2(cs, sn);
    float s1, c1;
    sincosf((float)CH * inv, &s1, &c1);
    sRot128[t] = make_float2(c1, s1);
  }
  int r = t >> 1, q = t & 1;
#pragma unroll
  for (int j = 0; j < 16; ++j) {
    int d = q * 16 + j;
    float inv = expf(-(float)d * ROPE_LC);
    sincosf((float)r * inv, &sr[j], &cr[j]);     // one-time, arg < 128
  }
}

__device__ inline void rope_advance(float2* sRotBase, const float2* sRot128) {
  int t = threadIdx.x;
  if (t < 32) {
    float2 b = sRotBase[t], r1 = sRot128[t];
    sRotBase[t] = make_float2(b.x * r1.x - b.y * r1.y,
                              b.y * r1.x + b.x * r1.y);
  }
}

// Prefetch one row of each listed tensor for chunk c (row r = tid>>1).
__device__ inline void prefetch_row(const float* p, size_t off) {
  __builtin_prefetch(p + off, 0, 3);        // global_prefetch_b8, first 128B
  __builtin_prefetch(p + off + 32, 0, 3);   // second 128B of the 256B row
}

// Load one chunk: rope(K)->sKt (and optionally sK row-major), rope(Q)->sQ
// (optional), V->sVt.  256 threads: thread t handles row r=t>>1, pair-half q=t&1.
__device__ inline void load_chunk(const float* __restrict__ Qb,
                                  const float* __restrict__ Kb,
                                  const float* __restrict__ Vb, int c,
                                  const float* cr, const float* sr,
                                  const float2* sRotBase,
                                  __bf16* sQ, __bf16* sK, __bf16* sKt,
                                  __bf16* sVt) {
  int t = threadIdx.x, r = t >> 1, q = t & 1;
  const float* krow = Kb + (size_t)(c * CH + r) * D;
  const float* vrow = Vb + (size_t)(c * CH + r) * D;
  const float* qrow = Qb ? (Qb + (size_t)(c * CH + r) * D) : nullptr;
#pragma unroll
  for (int j = 0; j < 16; ++j) {
    int d = q * 16 + j;
    float2 b0 = sRotBase[d];
    float cs = b0.x * cr[j] - b0.y * sr[j];
    float sn = b0.y * cr[j] + b0.x * sr[j];
    float b1 = krow[d], b2 = krow[d + 32];
    __bf16 kb1 = tobf(b1 * cs - b2 * sn);
    __bf16 kb2 = tobf(b1 * sn + b2 * cs);
    sKt[d * CH + r] = kb1;
    sKt[(d + 32) * CH + r] = kb2;
    if (sK) { sK[r * D + d] = kb1; sK[r * D + d + 32] = kb2; }
    if (sQ) {
      float a1 = qrow[d], a2 = qrow[d + 32];
      sQ[r * D + d]      = tobf(a1 * cs - a2 * sn);
      sQ[r * D + d + 32] = tobf(a1 * sn + a2 * cs);
    }
  }
#pragma unroll
  for (int i = 0; i < 8; ++i) {
    int d = q * 32 + i * 4;
    float4 vv = *(const float4*)(vrow + d);
    sVt[(d + 0) * CH + r] = tobf(vv.x);
    sVt[(d + 1) * CH + r] = tobf(vv.y);
    sVt[(d + 2) * CH + r] = tobf(vv.z);
    sVt[(d + 3) * CH + r] = tobf(vv.w);
  }
}

// ---------------- Kernel 1: per-segment KV sums: T[e][d] = sum_c V[c][e]*Kr[c][d]
__global__ __launch_bounds__(256)
void la_kv_kernel(const float* __restrict__ K, const float* __restrict__ V,
                  float* __restrict__ states, int G, int segChunks) {
  __shared__ __align__(16) __bf16 sKt[D * CH];
  __shared__ __align__(16) __bf16 sVt[D * CH];
  __shared__ float2 sRotBase[32];
  __shared__ float2 sRot128[32];
  int bh = blockIdx.x / G, g = blockIdx.x % G;
  const float* Kb = K + (size_t)bh * SSEQ * D;
  const float* Vb = V + (size_t)bh * SSEQ * D;
  int c0 = g * segChunks;
  int w = threadIdx.x >> 5, lane = threadIdx.x & 31;
  int nCol = lane & 15, mh = (lane >> 4) * 8;
  int et = w >> 1, dt0 = (w & 1) * 2;   // each wave owns 2 of 16 state tiles
  int r = threadIdx.x >> 1;

  float cr[16], sr[16];
  rope_init(c0, cr, sr, sRotBase, sRot128);
  __syncthreads();

  v8f accT[2] = {};
  for (int cc = 0; cc < segChunks; ++cc) {
    load_chunk(nullptr, Kb, Vb, c0 + cc, cr, sr, sRotBase,
               nullptr, nullptr, sKt, sVt);
    __syncthreads();
    rope_advance(sRotBase, sRot128);
    if (cc + 1 < segChunks) {           // warm L2 for next chunk during compute
      size_t off = (size_t)((c0 + cc + 1) * CH + r) * D;
      prefetch_row(Kb, off);
      prefetch_row(Vb, off);
    }
#pragma unroll
    for (int ks = 0; ks < 4; ++ks) {
      v16bf a = frag_a(sVt, CH, et * 16, ks * 32);
#pragma unroll
      for (int tt = 0; tt < 2; ++tt) {
        v16bf b = frag_bt(sKt, CH, (dt0 + tt) * 16, ks * 32);
        accT[tt] = wmma_bf16(a, b, accT[tt]);
      }
    }
    __syncthreads();
  }
  float* sp = states + (size_t)blockIdx.x * (D * D);
#pragma unroll
  for (int tt = 0; tt < 2; ++tt)
#pragma unroll
    for (int rr = 0; rr < 8; ++rr)
      sp[(et * 16 + mh + rr) * D + (dt0 + tt) * 16 + nCol] = accT[tt][rr];
}

// ---------------- Kernel 2: exclusive prefix over segments, in place
__global__ __launch_bounds__(256)
void la_prefix_kernel(float* __restrict__ states, int G) {
  int bh = blockIdx.x, t = threadIdx.x;
  float run[16];
#pragma unroll
  for (int j = 0; j < 16; ++j) run[j] = 0.f;
  for (int g = 0; g < G; ++g) {
    float* p = states + ((size_t)(bh * G + g)) * (D * D);
#pragma unroll
    for (int j = 0; j < 16; ++j) {
      int idx = j * 256 + t;
      float v = p[idx];
      p[idx] = run[j];
      run[j] += v;
    }
  }
}

// ---------------- Kernel 3: outputs for one segment (or whole seq if states==null)
__global__ __launch_bounds__(256)
void la_out_kernel(const float* __restrict__ Q, const float* __restrict__ K,
                   const float* __restrict__ V, float* __restrict__ Out,
                   const float* __restrict__ states, int G, int segChunks) {
  __shared__ __align__(16) __bf16 sQ[CH * D];
  __shared__ __align__(16) __bf16 sK[CH * D];
  __shared__ __align__(16) __bf16 sKt[D * CH];
  __shared__ __align__(16) __bf16 sVt[D * CH];
  __shared__ __align__(16) __bf16 sS[CH * CH];
  __shared__ __align__(16) float  sT[D * D];   // state^T (E x D), fp32 master
  __shared__ __align__(16) __bf16 sTb[D * D];  // bf16 shadow for WMMA B operand
  __shared__ float2 sRotBase[32];
  __shared__ float2 sRot128[32];

  int bh = blockIdx.x / G, g = blockIdx.x % G;
  size_t base = (size_t)bh * SSEQ * D;
  int c0 = g * segChunks;
  int t = threadIdx.x, w = t >> 5, lane = t & 31;
  int nCol = lane & 15, mh = (lane >> 4) * 8;
  int et = w >> 1, dt0 = (w & 1) * 2;
  int r = t >> 1;
  int wu = __builtin_amdgcn_readfirstlane(w);      // provably wave-uniform
  int ksMax = (16 * wu + 14) >> 5;                 // last k-block with nonzero S
  int ntMax = (wu & 1) ? wu : (wu + 1);            // S tiles that must be spilled

  float cr[16], sr[16];
  rope_init(c0, cr, sr, sRotBase, sRot128);

  // initial state for this segment
  if (states) {
    const float* sp = states + (size_t)blockIdx.x * (D * D);
    for (int i = t; i < D * D; i += 256) { float v = sp[i]; sT[i] = v; sTb[i] = tobf(v); }
  } else {
    for (int i = t; i < D * D; i += 256) { sT[i] = 0.f; sTb[i] = tobf(0.f); }
  }
  __syncthreads();

  for (int cc = 0; cc < segChunks; ++cc) {
    int c = c0 + cc;
    load_chunk(Q + base, K + base, V + base, c, cr, sr, sRotBase,
               sQ, sK, sKt, sVt);
    __syncthreads();
    rope_advance(sRotBase, sRot128);
    if (cc + 1 < segChunks) {           // warm L2 for next chunk during compute
      size_t off = (size_t)((c + 1) * CH + r) * D;
      prefetch_row(Q + base, off);
      prefetch_row(K + base, off);
      prefetch_row(V + base, off);
    }

    // ---- S = Qr * Kr^T : wave w owns row strip [16w,16w+16); tiles nt>w are
    //      fully below-diagonal-masked -> skipped (scalar branch on wu).
    v8f accS[8] = {};
#pragma unroll
    for (int ks = 0; ks < 2; ++ks) {
      v16bf a = frag_a(sQ, D, w * 16, ks * 32);
#pragma unroll
      for (int nt = 0; nt < 8; ++nt) {
        if (nt <= wu) {
          v16bf b = frag_bt(sK, D, nt * 16, ks * 32);
          accS[nt] = wmma_bf16(a, b, accS[nt]);
        }
      }
    }
    // strict lower-triangular mask (keep i>j), spill as bf16 (wave-local rows)
#pragma unroll
    for (int nt = 0; nt < 8; ++nt) {
      if (nt <= ntMax) {
#pragma unroll
        for (int rr = 0; rr < 8; ++rr) {
          int i = w * 16 + mh + rr, j = nt * 16 + nCol;
          sS[i * CH + j] = tobf(i > j ? accS[nt][rr] : 0.f);
        }
      }
    }

    // ---- out = maskedS * V (k-blocks beyond ksMax are all-zero)  +  Q * state
    v8f accO[4] = {};
#pragma unroll
    for (int ks = 0; ks < 4; ++ks) {
      if (ks <= ksMax) {
        v16bf a = frag_a(sS, CH, w * 16, ks * 32);
#pragma unroll
        for (int e = 0; e < 4; ++e) {
          v16bf b = frag_bt(sVt, CH, e * 16, ks * 32);
          accO[e] = wmma_bf16(a, b, accO[e]);
        }
      }
    }
#pragma unroll
    for (int ks = 0; ks < 2; ++ks) {
      v16bf a = frag_a(sQ, D, w * 16, ks * 32);
#pragma unroll
      for (int e = 0; e < 4; ++e) {
        v16bf b = frag_bt(sTb, D, e * 16, ks * 32);  // Bt rows = state^T rows
        accO[e] = wmma_bf16(a, b, accO[e]);
      }
    }
    float* orow = Out + base + (size_t)(c * CH) * D;
#pragma unroll
    for (int e = 0; e < 4; ++e)
#pragma unroll
      for (int rr = 0; rr < 8; ++rr)
        orow[(size_t)(w * 16 + mh + rr) * D + e * 16 + nCol] = accO[e][rr];

    // ---- state^T += V^T * Kr (skip after last chunk of segment)
    if (cc + 1 < segChunks) {
      v8f accT[2] = {};
#pragma unroll
      for (int ks = 0; ks < 4; ++ks) {
        v16bf a = frag_a(sVt, CH, et * 16, ks * 32);
#pragma unroll
        for (int tt = 0; tt < 2; ++tt) {
          v16bf b = frag_bt(sKt, CH, (dt0 + tt) * 16, ks * 32);
          accT[tt] = wmma_bf16(a, b, accT[tt]);
        }
      }
#pragma unroll
      for (int tt = 0; tt < 2; ++tt)
#pragma unroll
        for (int rr = 0; rr < 8; ++rr)
          sT[(et * 16 + mh + rr) * D + (dt0 + tt) * 16 + nCol] += accT[tt][rr];
      __syncthreads();
      for (int i = t; i < D * D; i += 256) sTb[i] = tobf(sT[i]);
      __syncthreads();
    }
  }
}

extern "C" void kernel_launch(void* const* d_in, const int* in_sizes, int n_in,
                              void* d_out, int out_size, void* d_ws, size_t ws_size,
                              hipStream_t stream) {
  (void)in_sizes; (void)n_in; (void)out_size;
  const float* Q = (const float*)d_in[0];
  const float* K = (const float*)d_in[1];
  const float* V = (const float*)d_in[2];
  float* Out = (float*)d_out;

  size_t need = (size_t)BH * SEGG * D * D * sizeof(float);
  if (d_ws && ws_size >= need) {
    float* st = (float*)d_ws;
    la_kv_kernel<<<BH * SEGG, 256, 0, stream>>>(K, V, st, SEGG, SEGC);
    la_prefix_kernel<<<BH, 256, 0, stream>>>(st, SEGG);
    la_out_kernel<<<BH * SEGG, 256, 0, stream>>>(Q, K, V, Out, st, SEGG, SEGC);
  } else {
    // fallback: 64 persistent workgroups, full sequential scan, no scratch
    la_out_kernel<<<BH, 256, 0, stream>>>(Q, K, V, Out, nullptr, 1, NCH);
  }
}
